// HashedLinear_11501922419140
// MI455X (gfx1250) — matrix operand order repaired and checked by reading, not compiled, not run on hardware
//
#include <hip/hip_runtime.h>

// HashedLinear: out[1024,4096] = x[1024,4096] @ w[indx[4096,4096]] + b[4096]
// Phase 1: convert x to f16, gather+transpose shared weights to f16 WT[n][k].
// Phase 2: f16 WMMA GEMM with f32 accumulation, async global->LDS double buffering.

#ifndef __has_builtin
#define __has_builtin(x) 0
#endif

#if defined(__HIP_DEVICE_COMPILE__) && \
    __has_builtin(__builtin_amdgcn_global_load_async_to_lds_b128)
#define HAVE_ASYNC 1
#else
#define HAVE_ASYNC 0
#endif

typedef int v4i __attribute__((vector_size(16)));

#if HAVE_ASYNC
#if __has_builtin(__builtin_amdgcn_s_wait_asynccnt)
#define WAIT_ASYNC(n) __builtin_amdgcn_s_wait_asynccnt(n)
#else
#define WAIT_ASYNC(n) asm volatile("s_wait_asynccnt " #n ::: "memory")
#endif
// copy 16B per lane: global -> LDS, no VGPR staging, tracked by ASYNCcnt.
#define ASYNC_CP16(gp, lp)                                        \
    __builtin_amdgcn_global_load_async_to_lds_b128(               \
        (__attribute__((address_space(1))) v4i*)(gp),             \
        (__attribute__((address_space(3))) v4i*)(lp), 0, 0)
#endif

typedef __attribute__((ext_vector_type(16))) _Float16 v16h;
typedef __attribute__((ext_vector_type(8)))  _Float16 v8h;
typedef __attribute__((ext_vector_type(8)))  float    v8f;

#define M_DIM 1024
#define K_DIM 4096
#define N_DIM 4096

union FragU { v16h v; v8h h[2]; };

// ---------------- Phase 1a: x fp32 -> f16 ----------------
__global__ void __launch_bounds__(256) cvt_x_kernel(const float* __restrict__ x,
                                                    _Float16* __restrict__ xh) {
    int i = (blockIdx.x * 256 + threadIdx.x) * 8;
    float4 a = *(const float4*)(x + i);
    float4 b = *(const float4*)(x + i + 4);
    v8h o;
    o[0] = (_Float16)a.x; o[1] = (_Float16)a.y; o[2] = (_Float16)a.z; o[3] = (_Float16)a.w;
    o[4] = (_Float16)b.x; o[5] = (_Float16)b.y; o[6] = (_Float16)b.z; o[7] = (_Float16)b.w;
    *(v8h*)(xh + i) = o;
}

// ---------------- Phase 1b: WT[n][k] = (f16) w[indx[k][n]] (gather + transpose) ----------------
__global__ void __launch_bounds__(256) gather_wt_kernel(const float* __restrict__ w,
                                                        const int* __restrict__ indx,
                                                        _Float16* __restrict__ wt) {
    __shared__ _Float16 tile[64][72];   // 64x64 tile, padded rows
    const int n0 = blockIdx.x * 64;
    const int k0 = blockIdx.y * 64;
    const int t  = threadIdx.x;
    const int nl = t & 63;              // coalesced along n for indx reads
    const int kb = (t >> 6) * 16;
    #pragma unroll
    for (int r = 0; r < 16; ++r) {
        int kl  = kb + r;
        int idx = indx[(size_t)(k0 + kl) * N_DIM + (n0 + nl)];
        tile[kl][nl] = (_Float16)w[idx];   // w pool (256KB) is L2-resident
    }
    __syncthreads();
    const int kc = (t & 7) * 8;         // 16B-vector writes, coalesced along k
    const int nb = t >> 3;              // 0..31
    #pragma unroll
    for (int h = 0; h < 2; ++h) {
        int nl2 = nb + h * 32;
        v8h o;
        #pragma unroll
        for (int ii = 0; ii < 8; ++ii) o[ii] = tile[kc + ii][nl2];
        *(v8h*)(wt + (size_t)(n0 + nl2) * K_DIM + (k0 + kc)) = o;
    }
}

// ---------------- Phase 2: GEMM via v_wmma_f32_16x16x32_f16 ----------------
#define BM 128
#define BN 128
#define BK 64
#define LDSP (BK + 8)   // pad to stagger LDS banks

__global__ void __launch_bounds__(256) hashed_gemm_kernel(
    const _Float16* __restrict__ xh,   // [M][K] row-major
    const _Float16* __restrict__ wt,   // [N][K] row-major (pre-transposed)
    const float*    __restrict__ bias, // [N]
    float*          __restrict__ out)  // [M][N]
{
    __shared__ _Float16 sA[2][BM][LDSP];   // 36864 B
    __shared__ _Float16 sB[2][BN][LDSP];   // 36864 B  (73.7 KB total, < 320 KB/WGP)

    const int t    = threadIdx.x;
    const int lane = t & 31;
    const int wv   = t >> 5;       // 8 waves
    const int wm   = wv >> 2;      // 0..1 : 64-row patch
    const int wn   = wv & 3;       // 0..3 : 32-col patch
    const int m0   = blockIdx.y * BM;
    const int n0   = blockIdx.x * BN;

    const int hi     = lane >> 4;  // lane half-select
    const int l15    = lane & 15;
    const int a_koff = hi * 8;     // A frag: K {0..7,16..23} / {8..15,24..31}
    const int b_koff = hi * 8;     // B frag: K {0..7,16..23} / {8..15,24..31}

    v8f zero = {};
    v8f acc[4][2];                 // 4 M-frags x 2 N-frags per wave (64x32)
    #pragma unroll
    for (int i = 0; i < 4; ++i)
        #pragma unroll
        for (int j = 0; j < 2; ++j) acc[i][j] = zero;

    auto compute = [&](int buf) {
        #pragma unroll
        for (int ks = 0; ks < BK; ks += 32) {
            FragU af[4], bf[2];
            #pragma unroll
            for (int i = 0; i < 4; ++i) {
                int row = wm * 64 + i * 16 + l15;
                const _Float16* p = &sA[buf][row][ks + a_koff];
                af[i].h[0] = *(const v8h*)p;
                af[i].h[1] = *(const v8h*)(p + 16);
            }
            #pragma unroll
            for (int j = 0; j < 2; ++j) {
                int row = wn * 32 + j * 16 + l15;
                const _Float16* p = &sB[buf][row][ks + b_koff];
                bf[j].h[0] = *(const v8h*)p;
                bf[j].h[1] = *(const v8h*)(p + 16);
            }
            #pragma unroll
            for (int i = 0; i < 4; ++i)
                #pragma unroll
                for (int j = 0; j < 2; ++j)
                    acc[i][j] = __builtin_amdgcn_wmma_f32_16x16x32_f16(
                        false, af[i].v, false, bf[j].v,
                        (short)0, acc[i][j], false, false);
        }
    };

    const int T = K_DIM / BK;   // 64

#if HAVE_ASYNC
    // ---- async global->LDS pipeline: zero VGPR staging, ASYNCcnt tracked ----
    auto issue = [&](int kt, int buf) {
        #pragma unroll
        for (int i = 0; i < 4; ++i) {   // A tile: 1024 x 16B chunks / 256 thr
            int c = t + i * 256, row = c >> 3, col = (c & 7) * 8;
            ASYNC_CP16(xh + (size_t)(m0 + row) * K_DIM + kt + col,
                       &sA[buf][row][col]);
        }
        #pragma unroll
        for (int i = 0; i < 4; ++i) {   // B tile: 1024 x 16B chunks / 256 thr
            int c = t + i * 256, row = c >> 3, col = (c & 7) * 8;
            ASYNC_CP16(wt + (size_t)(n0 + row) * K_DIM + kt + col,
                       &sB[buf][row][col]);
        }
    };

    issue(0, 0);
    for (int ti = 0; ti < T; ++ti) {
        int cur = ti & 1;
        if (ti + 1 < T) {
            issue((ti + 1) * BK, cur ^ 1);  // prefetch next stage
            WAIT_ASYNC(8);                  // first 8 (current stage) complete
        } else {
            WAIT_ASYNC(0);
        }
        __syncthreads();                    // all waves' writes visible
        compute(cur);
        __syncthreads();                    // done reading before next overwrite
    }
#else
    // ---- fallback: reg-staged copy, loads and stores adjacent (no spill) ----
    uint4 ra[4], rb[4];
    for (int ti = 0; ti < T; ++ti) {
        int cur = ti & 1;
        int kt = ti * BK;
        #pragma unroll
        for (int i = 0; i < 4; ++i) {
            int c = t + i * 256, row = c >> 3, col = (c & 7) * 8;
            ra[i] = *(const uint4*)(xh + (size_t)(m0 + row) * K_DIM + kt + col);
        }
        #pragma unroll
        for (int i = 0; i < 4; ++i) {
            int c = t + i * 256, row = c >> 3, col = (c & 7) * 8;
            rb[i] = *(const uint4*)(wt + (size_t)(n0 + row) * K_DIM + kt + col);
        }
        #pragma unroll
        for (int i = 0; i < 4; ++i) {
            int c = t + i * 256, row = c >> 3, col = (c & 7) * 8;
            *(uint4*)&sA[cur][row][col] = ra[i];
        }
        #pragma unroll
        for (int i = 0; i < 4; ++i) {
            int c = t + i * 256, row = c >> 3, col = (c & 7) * 8;
            *(uint4*)&sB[cur][row][col] = rb[i];
        }
        __syncthreads();
        compute(cur);
        __syncthreads();
    }
#endif

    // epilogue: C/D layout -> VGPR r: lanes0-15 M=r, lanes16-31 M=r+8; N=lane&15
    #pragma unroll
    for (int j = 0; j < 2; ++j) {
        int n = n0 + wn * 32 + j * 16 + l15;
        float bv = bias[n];
        #pragma unroll
        for (int i = 0; i < 4; ++i) {
            int mbase = m0 + wm * 64 + i * 16 + hi * 8;
            #pragma unroll
            for (int r = 0; r < 8; ++r)
                out[(size_t)(mbase + r) * N_DIM + n] = acc[i][j][r] + bv;
        }
    }
}

extern "C" void kernel_launch(void* const* d_in, const int* in_sizes, int n_in,
                              void* d_out, int out_size, void* d_ws, size_t ws_size,
                              hipStream_t stream) {
    const float* x    = (const float*)d_in[0];
    const float* w    = (const float*)d_in[1];
    const float* b    = (const float*)d_in[2];
    const int*   indx = (const int*)d_in[3];
    float*       out  = (float*)d_out;

    _Float16* xh = (_Float16*)d_ws;                                      // 8 MB
    _Float16* wt = (_Float16*)((char*)d_ws + (size_t)M_DIM * K_DIM * 2); // +32 MB

    cvt_x_kernel<<<(M_DIM * K_DIM) / (256 * 8), 256, 0, stream>>>(x, xh);
    gather_wt_kernel<<<dim3(N_DIM / 64, K_DIM / 64), 256, 0, stream>>>(w, indx, wt);
    hashed_gemm_kernel<<<dim3(N_DIM / BN, M_DIM / BM), 256, 0, stream>>>(xh, wt, b, out);
}
